// GT_Cos_17497696764591
// MI455X (gfx1250) — matrix operand order repaired and checked by reading, not compiled, not run on hardware
//
#include <hip/hip_runtime.h>
#include <hip/hip_bf16.h>

#define NROWS 8192
#define INDIM 256
#define HIDD  128

typedef __bf16 bf16_t;
typedef __attribute__((ext_vector_type(16))) __bf16 v16bf;
typedef __attribute__((ext_vector_type(8)))  __bf16 v8bf;
typedef __attribute__((ext_vector_type(8)))  float  v8f;

__device__ __forceinline__ v8f wmma_bf16(v16bf a, v16bf b, v8f c) {
  // D = A(16x32 bf16) x B(32x16 bf16) + C(16x16 f32)
  return __builtin_amdgcn_wmma_f32_16x16x32_bf16(false, a, false, b, (short)0, c, false, false);
}

// ---- fragment loaders -------------------------------------------------------
// A fragment (16x32): lane<16 -> row=lane, K {0..7, 16..23}; lane>=16 -> K {8..15, 24..31}
__device__ __forceinline__ v16bf load_a_bf16(const bf16_t* A, int lda, int row0, int k0, int lane) {
  const int m  = lane & 15;
  const int ko = (lane & 16) ? 8 : 0;
  const bf16_t* p = A + (size_t)(row0 + m) * lda + (k0 + ko);
  v8bf lo = *(const v8bf*)p;
  v8bf hi = *(const v8bf*)(p + 16);
  v16bf r;
#pragma unroll
  for (int i = 0; i < 8; ++i) { r[i] = lo[i]; r[i + 8] = hi[i]; }
  return r;
}

__device__ __forceinline__ v16bf load_a_f32(const float* A, int lda, int row0, int k0, int lane) {
  const int m  = lane & 15;
  const int ko = (lane & 16) ? 8 : 0;
  const float* p = A + (size_t)(row0 + m) * lda + (k0 + ko);
  v16bf r;
#pragma unroll
  for (int i = 0; i < 8; ++i) { r[i] = (bf16_t)p[i]; r[i + 8] = (bf16_t)p[i + 16]; }
  return r;
}

// B fragment (32x16) from Bt row-major [ncols, ldb] where Bt[n,k] == B[k,n]
// lane<16 -> col=lane,    K {0..7, 16..23}; lane>=16 -> col=lane-16, K {8..15, 24..31}
__device__ __forceinline__ v16bf load_b_t(const bf16_t* Bt, int ldb, int n0, int k0, int lane) {
  const int n  = lane & 15;
  const int ko = (lane & 16) ? 8 : 0;
  const bf16_t* p = Bt + (size_t)(n0 + n) * ldb + (k0 + ko);
  v8bf lo = *(const v8bf*)p;
  v8bf hi = *(const v8bf*)(p + 16);
  v16bf r;
#pragma unroll
  for (int i = 0; i < 8; ++i) { r[i] = lo[i]; r[i + 8] = hi[i]; }
  return r;
}

// ---- tiny weight transpose: W[K,D] f32 -> Wt[D,K] bf16 ----------------------
__global__ void transpose_w_kernel(const float* __restrict__ W, bf16_t* __restrict__ Wt,
                                   int K, int D) {
  int idx = blockIdx.x * blockDim.x + threadIdx.x;
  if (idx < K * D) {
    int k = idx / D, d = idx - k * D;
    Wt[(size_t)d * K + k] = (bf16_t)W[(size_t)k * D + d];
  }
}

// ---- projection GEMM: out = A[NROWS,KDIM] @ W[KDIM,Dout] + bias -------------
// MODE 0: bf16 row-major out [NROWS,Dout]
// MODE 1: bf16 transposed out [Dout,NROWS]   (for V^T)
// MODE 2: f32 row-major out  [NROWS,Dout]    (skip connection)
template<int KDIM, bool ABF16, int MODE>
__global__ __launch_bounds__(128) void proj_kernel(const void* __restrict__ Av,
                                                   const bf16_t* __restrict__ Wt,
                                                   const float* __restrict__ bias,
                                                   void* __restrict__ outv, int Dout) {
  const int lane = threadIdx.x & 31;
  const int wave = threadIdx.x >> 5;
  const int tile = blockIdx.x * 4 + wave;
  const int tcols = Dout >> 4;
  const int ti = tile / tcols;
  const int tj = tile - ti * tcols;

  v8f acc = {};
#pragma unroll
  for (int kb = 0; kb < KDIM; kb += 32) {
    v16bf a;
    if (ABF16) a = load_a_bf16((const bf16_t*)Av, KDIM, ti * 16, kb, lane);
    else       a = load_a_f32 ((const float*)Av,  KDIM, ti * 16, kb, lane);
    v16bf b = load_b_t(Wt, KDIM, tj * 16, kb, lane);
    acc = wmma_bf16(a, b, acc);
  }
  const int col  = tj * 16 + (lane & 15);
  const float bv = bias[col];
  const int mrow = ti * 16 + ((lane & 16) ? 8 : 0);
#pragma unroll
  for (int r = 0; r < 8; ++r) {
    float v = acc[r] + bv;
    if (MODE == 0)      ((bf16_t*)outv)[(size_t)(mrow + r) * Dout + col] = (bf16_t)v;
    else if (MODE == 1) ((bf16_t*)outv)[(size_t)col * NROWS + (mrow + r)] = (bf16_t)v;
    else                ((float*)outv)[(size_t)(mrow + r) * Dout + col] = v;
  }
}

// ---- flash attention: out = softmax(Q K^T * scale) V + skip -----------------
// Q,K: bf16 [NROWS,D] row-major; Vt: bf16 [D,NROWS]; skip: f32 [NROWS,D]
template<int D, bool DO_ELU, bool OUT_BF16>
__global__ __launch_bounds__(128) void flash_kernel(const bf16_t* __restrict__ Q,
                                                    const bf16_t* __restrict__ Km,
                                                    const bf16_t* __restrict__ Vt,
                                                    const float* __restrict__ skip,
                                                    void* __restrict__ outv, float scale) {
  __shared__ __align__(16) bf16_t pbuf[4][16 * 32];
  const int lane = threadIdx.x & 31;
  const int wave = threadIdx.x >> 5;
  const int ti   = blockIdx.x * 4 + wave;
  const int row0 = ti * 16;

  v8f zero = {};
  v16bf qf[D / 32];
#pragma unroll
  for (int c = 0; c < D / 32; ++c) qf[c] = load_a_bf16(Q, D, row0, c * 32, lane);

  v8f of[D / 16];
#pragma unroll
  for (int c = 0; c < D / 16; ++c) of[c] = zero;

  float mrow[8], lrow[8];
#pragma unroll
  for (int r = 0; r < 8; ++r) { mrow[r] = -1e30f; lrow[r] = 0.0f; }

  bf16_t* pb = pbuf[wave];
  const int cn = lane & 15;
  const int mo = (lane & 16) ? 8 : 0;

  for (int j = 0; j < NROWS; j += 32) {
    // S tiles for key halves [j, j+16) and [j+16, j+32)
    v8f s0 = zero, s1 = zero;
#pragma unroll
    for (int c = 0; c < D / 32; ++c) s0 = wmma_bf16(qf[c], load_b_t(Km, D, j,      c * 32, lane), s0);
#pragma unroll
    for (int c = 0; c < D / 32; ++c) s1 = wmma_bf16(qf[c], load_b_t(Km, D, j + 16, c * 32, lane), s1);

    // online softmax stats (row max / rescale factor) via 16-lane reductions
    float fac[8], nm[8];
#pragma unroll
    for (int r = 0; r < 8; ++r) {
      float a = s0[r] * scale, b = s1[r] * scale;
      s0[r] = a; s1[r] = b;
      float mx = fmaxf(a, b);
      mx = fmaxf(mx, __shfl_xor(mx, 1, 32));
      mx = fmaxf(mx, __shfl_xor(mx, 2, 32));
      mx = fmaxf(mx, __shfl_xor(mx, 4, 32));
      mx = fmaxf(mx, __shfl_xor(mx, 8, 32));
      float nmx = fmaxf(mrow[r], mx);
      fac[r] = __expf(mrow[r] - nmx);
      mrow[r] = nmx;
      nm[r] = nmx;
    }
    // exponentiate, row-sum, stash P (bf16) into LDS in [16 x 32] tile form
#pragma unroll
    for (int r = 0; r < 8; ++r) {
      float p0 = __expf(s0[r] - nm[r]);
      float p1 = __expf(s1[r] - nm[r]);
      float rs = p0 + p1;
      rs += __shfl_xor(rs, 1, 32);
      rs += __shfl_xor(rs, 2, 32);
      rs += __shfl_xor(rs, 4, 32);
      rs += __shfl_xor(rs, 8, 32);
      lrow[r] = lrow[r] * fac[r] + rs;
      pb[(r + mo) * 32 + cn]      = (bf16_t)p0;
      pb[(r + mo) * 32 + 16 + cn] = (bf16_t)p1;
    }
    // same-wave DS ops are in-order; wait + compiler barrier before re-read
    asm volatile("s_wait_dscnt 0" ::: "memory");
    // reload P in A-matrix layout
    v16bf pf;
    {
      const int m  = lane & 15;
      const int ko = (lane & 16) ? 8 : 0;
      v8bf lo = *(const v8bf*)&pb[m * 32 + ko];
      v8bf hi = *(const v8bf*)&pb[m * 32 + ko + 16];
#pragma unroll
      for (int i = 0; i < 8; ++i) { pf[i] = lo[i]; pf[i + 8] = hi[i]; }
    }
    // rescale running O, then O += P @ V  (V^T rows give B fragments directly)
#pragma unroll
    for (int c = 0; c < D / 16; ++c) {
#pragma unroll
      for (int r = 0; r < 8; ++r) of[c][r] *= fac[r];
      of[c] = wmma_bf16(pf, load_b_t(Vt, NROWS, c * 16, j, lane), of[c]);
    }
    if (j + 32 < NROWS) {
      __builtin_prefetch(Km + (size_t)(j + 32) * D, 0, 1);
      __builtin_prefetch(Vt + (j + 32), 0, 1);
    }
  }

  // finalize: O/l + skip (+ELU), store
  const int mbase = row0 + mo;
#pragma unroll
  for (int c = 0; c < D / 16; ++c) {
    const int col = c * 16 + cn;
#pragma unroll
    for (int r = 0; r < 8; ++r) {
      float v = of[c][r] / lrow[r] + skip[(size_t)(mbase + r) * D + col];
      if (DO_ELU) v = (v > 0.0f) ? v : (__expf(v) - 1.0f);
      if (OUT_BF16) ((bf16_t*)outv)[(size_t)(mbase + r) * D + col] = (bf16_t)v;
      else          ((float*)outv)[(size_t)(mbase + r) * D + col] = v;
    }
  }
}

// ---- driver -----------------------------------------------------------------
extern "C" void kernel_launch(void* const* d_in, const int* in_sizes, int n_in,
                              void* d_out, int out_size, void* d_ws, size_t ws_size,
                              hipStream_t stream) {
  (void)in_sizes; (void)n_in; (void)out_size; (void)ws_size;
  const float* x   = (const float*)d_in[0];
  // d_in[1]=XY_Adj, d_in[2]=w0, d_in[3]=w1 are dead w.r.t. the output (sim unused)
  const float* Wq1 = (const float*)d_in[4];  const float* bq1 = (const float*)d_in[5];
  const float* Wk1 = (const float*)d_in[6];  const float* bk1 = (const float*)d_in[7];
  const float* Wv1 = (const float*)d_in[8];  const float* bv1 = (const float*)d_in[9];
  const float* Ws1 = (const float*)d_in[10]; const float* bs1 = (const float*)d_in[11];
  const float* Wq4 = (const float*)d_in[12]; const float* bq4 = (const float*)d_in[13];
  const float* Wk4 = (const float*)d_in[14]; const float* bk4 = (const float*)d_in[15];
  const float* Wv4 = (const float*)d_in[16]; const float* bv4 = (const float*)d_in[17];
  const float* Ws4 = (const float*)d_in[18]; const float* bs4 = (const float*)d_in[19];

  uint8_t* ws = (uint8_t*)d_ws;
  size_t off = 0;
  auto carve = [&](size_t bytes) -> void* {
    void* p = ws + off;
    off += (bytes + 255) & ~(size_t)255;
    return p;
  };
  const size_t WE = (size_t)INDIM * HIDD;              // elements per weight matrix
  bf16_t* Wq1t = (bf16_t*)carve(WE * 2);
  bf16_t* Wk1t = (bf16_t*)carve(WE * 2);
  bf16_t* Wv1t = (bf16_t*)carve(WE * 2);
  bf16_t* Ws1t = (bf16_t*)carve(WE * 2);
  bf16_t* Wq4t = (bf16_t*)carve(WE * 2);
  bf16_t* Wk4t = (bf16_t*)carve(WE * 2);
  bf16_t* Wv4t = (bf16_t*)carve(WE * 2);
  bf16_t* Ws4t = (bf16_t*)carve(WE * 2);
  bf16_t* Q1  = (bf16_t*)carve((size_t)NROWS * HIDD * 2);
  bf16_t* K1  = (bf16_t*)carve((size_t)NROWS * HIDD * 2);
  bf16_t* V1t = (bf16_t*)carve((size_t)NROWS * HIDD * 2);
  float*  S1  = (float*) carve((size_t)NROWS * HIDD * 4);
  bf16_t* h1  = (bf16_t*)carve((size_t)NROWS * HIDD * 2);
  bf16_t* Q2  = (bf16_t*)carve((size_t)NROWS * INDIM * 2);
  bf16_t* K2  = (bf16_t*)carve((size_t)NROWS * INDIM * 2);
  bf16_t* V2t = (bf16_t*)carve((size_t)NROWS * INDIM * 2);
  float*  S2  = (float*) carve((size_t)NROWS * INDIM * 4);

  // weight transposes (f32 -> bf16), 32K elements each
  const int twBlocks = (INDIM * HIDD + 255) / 256;
  transpose_w_kernel<<<twBlocks, 256, 0, stream>>>(Wq1, Wq1t, INDIM, HIDD);
  transpose_w_kernel<<<twBlocks, 256, 0, stream>>>(Wk1, Wk1t, INDIM, HIDD);
  transpose_w_kernel<<<twBlocks, 256, 0, stream>>>(Wv1, Wv1t, INDIM, HIDD);
  transpose_w_kernel<<<twBlocks, 256, 0, stream>>>(Ws1, Ws1t, INDIM, HIDD);
  transpose_w_kernel<<<twBlocks, 256, 0, stream>>>(Wq4, Wq4t, HIDD, INDIM);
  transpose_w_kernel<<<twBlocks, 256, 0, stream>>>(Wk4, Wk4t, HIDD, INDIM);
  transpose_w_kernel<<<twBlocks, 256, 0, stream>>>(Wv4, Wv4t, HIDD, INDIM);
  transpose_w_kernel<<<twBlocks, 256, 0, stream>>>(Ws4, Ws4t, HIDD, INDIM);

  // layer 1 projections: x[8192,256] @ W[256,128]
  const int p1Blocks = (NROWS / 16) * (HIDD / 16) / 4;  // 1024
  proj_kernel<INDIM, false, 0><<<p1Blocks, 128, 0, stream>>>(x, Wq1t, bq1, Q1,  HIDD);
  proj_kernel<INDIM, false, 0><<<p1Blocks, 128, 0, stream>>>(x, Wk1t, bk1, K1,  HIDD);
  proj_kernel<INDIM, false, 1><<<p1Blocks, 128, 0, stream>>>(x, Wv1t, bv1, V1t, HIDD);
  proj_kernel<INDIM, false, 2><<<p1Blocks, 128, 0, stream>>>(x, Ws1t, bs1, S1,  HIDD);

  // layer 1 attention + skip + ELU -> h1 (bf16)
  flash_kernel<HIDD, true, true><<<NROWS / 16 / 4, 128, 0, stream>>>(
      Q1, K1, V1t, S1, h1, 0.08838834764831845f /* 1/sqrt(128) */);

  // layer 2 projections: h1[8192,128] @ W[128,256]
  const int p2Blocks = (NROWS / 16) * (INDIM / 16) / 4;  // 2048
  proj_kernel<HIDD, true, 0><<<p2Blocks, 128, 0, stream>>>(h1, Wq4t, bq4, Q2,  INDIM);
  proj_kernel<HIDD, true, 0><<<p2Blocks, 128, 0, stream>>>(h1, Wk4t, bk4, K2,  INDIM);
  proj_kernel<HIDD, true, 1><<<p2Blocks, 128, 0, stream>>>(h1, Wv4t, bv4, V2t, INDIM);
  proj_kernel<HIDD, true, 2><<<p2Blocks, 128, 0, stream>>>(h1, Ws4t, bs4, S2,  INDIM);

  // layer 2 attention + skip -> d_out (f32)
  flash_kernel<INDIM, false, false><<<NROWS / 16 / 4, 128, 0, stream>>>(
      Q2, K2, V2t, S2, d_out, 0.0625f /* 1/sqrt(256) */);
}
